// PConvLinearOpt_8967891714687
// MI455X (gfx1250) — matrix operand order repaired and checked by reading, not compiled, not run on hardware
//
#include <hip/hip_runtime.h>
#include <hip/hip_bf16.h>
#include <stdint.h>

typedef __attribute__((ext_vector_type(16))) _Float16 v16h;
typedef __attribute__((ext_vector_type(8)))  float    v8f;

#define BB      2
#define NN      40000
#define KK      16
#define C_IN    64
#define C_MID   16
#define C_ADD   3
#define C_OUT   128
#define C_TOT   67      // 64 + 3
#define C_TOT_P 80      // padded to 5 tiles of 16
#define F_REAL  1072    // 67*16
#define F_PAD   1088    // 34 tiles of 32
#define NKT     34      // K-tiles of 32 in the output GEMM
#define PTS     16      // points per workgroup
#define NWAVE   8

// ---- gfx1250 async Global->LDS path (guarded: falls back to plain loads) ----
#if defined(__has_builtin)
#  if __has_builtin(__builtin_amdgcn_global_load_async_to_lds_b128)
#    define USE_ASYNC 1
#  endif
#endif

#if defined(__has_builtin)
#  if __has_builtin(__builtin_amdgcn_s_wait_asynccnt)
#    define WAIT_ASYNC0() __builtin_amdgcn_s_wait_asynccnt(0)
#  else
#    define WAIT_ASYNC0() asm volatile("s_wait_asynccnt 0" ::: "memory")
#  endif
#else
#  define WAIT_ASYNC0() asm volatile("s_wait_asynccnt 0" ::: "memory")
#endif

// Builtin parameter types (from hipcc diagnostic: int __vector(4) in AS1/AS3)
typedef int v4i __attribute__((vector_size(16)));
typedef __attribute__((address_space(1))) v4i* gv4i_t;
typedef __attribute__((address_space(3))) v4i* lv4i_t;

// ---------------------------------------------------------------------------
// Prep kernel: repack linear_weights (f32 [128][1072]) into f16 WMMA
// B-fragment order in d_ws:  [kt][otile][lane][8 x uint(2 halves)]
// B 32x16 layout (wave32): VGPR v<4 -> K=8*(lane/16)+2v,+1; v>=4 -> K=16+8*(lane/16)+2(v-4),+1;
// N = lane % 16.
// ---------------------------------------------------------------------------
__global__ void prep_weights(const float* __restrict__ W, uint32_t* __restrict__ outw) {
    int u = blockIdx.x * blockDim.x + threadIdx.x;     // uint index
    int j2   = u & 7;            // uint within lane (halves 2*j2, 2*j2+1)
    int lane = (u >> 3) & 31;
    int w    = (u >> 8) & 7;     // output tile
    int kt   = u >> 11;
    if (kt >= NKT) return;
    int o     = w * 16 + (lane & 15);
    int kbase = ((j2 < 4) ? 0 : 16) + ((lane < 16) ? 0 : 8) + 2 * (j2 & 3);
    _Float16 h[2];
    #pragma unroll
    for (int d = 0; d < 2; ++d) {
        int f = kt * 32 + kbase + d;
        float val = (f < F_REAL) ? W[(size_t)o * F_REAL + f] : 0.0f;
        h[d] = (_Float16)val;
    }
    uint32_t packed;
    __builtin_memcpy(&packed, h, 4);
    outw[u] = packed;
}

// ---------------------------------------------------------------------------
// Fused kernel: gather -> pconv (WMMA) -> linear (WMMA)
// One workgroup = 256 threads = 8 waves = 16 points.
// ---------------------------------------------------------------------------
__global__ __launch_bounds__(256)
void pconv_fused(const float* __restrict__ input_feat,
                 const long long* __restrict__ ninds,
                 const float* __restrict__ weightnet,
                 const float* __restrict__ addfeat,
                 const float* __restrict__ bias,
                 const uint32_t* __restrict__ Wpk,
                 float* __restrict__ out) {
    __shared__ _Float16 featT[PTS][C_TOT_P][KK];   // 40960 B  feat transposed [p][c][k]
    __shared__ _Float16 wnT  [PTS][C_MID][KK];     //  8192 B  weightnet transposed [p][m][k]
    __shared__ _Float16 pconv[PTS][F_PAD];         // 34816 B  flattened pconv [p][c*16+m]
#ifdef USE_ASYNC
    __shared__ float    wn_raw[PTS * KK * C_MID];  // 16384 B  raw f32 weightnet slab
#endif

    const int tid  = threadIdx.x;
    const int lane = tid & 31;
    const int wv   = tid >> 5;
    const int g0   = blockIdx.x * PTS;             // first global point (b*N+n flat)

#ifdef USE_ASYNC
    // --- kick off async Global->LDS copy of the contiguous weightnet slab ---
    // slab = 16 points * 256 floats = 16 KB starting at weightnet + g0*256.
    {
        const float* base = weightnet + (size_t)g0 * (KK * C_MID);
        #pragma unroll
        for (int j = 0; j < 4; ++j) {
            const int i = tid + j * 256;           // 1024 x 16B transfers
            __builtin_amdgcn_global_load_async_to_lds_b128(
                (gv4i_t)(base + i * 4),
                (lv4i_t)(&wn_raw[i * 4]),
                0, 0);
        }
    }
#endif

    // ---------------- Phase 1: gather into LDS (f32 -> f16) ----------------
    {
        const int p = tid >> 4;                    // 0..15 point in tile
        const int k = tid & 15;                    // 0..15 neighbor
        const int g = g0 + p;
        const int b = g / NN;
        const int n = g % NN;
        const long long nb = ninds[((size_t)b * NN + n) * KK + k];
        const float* src = input_feat + ((size_t)b * NN + (size_t)nb) * C_IN;
        #pragma unroll
        for (int c = 0; c < C_IN; c += 4) {
            float4 q = *(const float4*)(src + c);
            featT[p][c + 0][k] = (_Float16)q.x;
            featT[p][c + 1][k] = (_Float16)q.y;
            featT[p][c + 2][k] = (_Float16)q.z;
            featT[p][c + 3][k] = (_Float16)q.w;
        }
        const float* af = addfeat + (((size_t)b * NN + n) * KK + k) * C_ADD;
        featT[p][64][k] = (_Float16)af[0];
        featT[p][65][k] = (_Float16)af[1];
        featT[p][66][k] = (_Float16)af[2];
        #pragma unroll
        for (int c = C_TOT; c < C_TOT_P; ++c) featT[p][c][k] = (_Float16)0.0f;

#ifdef USE_ASYNC
        WAIT_ASYNC0();                 // our async transfers done (LDS written)
        __syncthreads();               // all waves' transfers + feat visible
        {
            const float* wn = &wn_raw[(p * KK + k) * C_MID];
            #pragma unroll
            for (int m = 0; m < C_MID; ++m) wnT[p][m][k] = (_Float16)wn[m];
        }
#else
        const float* wn = weightnet + (((size_t)b * NN + n) * KK + k) * C_MID;
        #pragma unroll
        for (int m = 0; m < C_MID; ++m) wnT[p][m][k] = (_Float16)wn[m];
#endif
    }
    __syncthreads();

    // ------- Phase 2: pconv[c,m] = sum_k feat[k,c]*wn[k,m]  (WMMA, K pad 32) -------
    {
        const int hi = lane >> 4;                  // lane group
        const int lm = lane & 15;
        #pragma unroll
        for (int pi = 0; pi < 2; ++pi) {
            const int p = wv * 2 + pi;
            // B fragment: wn (K=0..15 real, 16..31 zero). lane half h, elem s<8 -> K=8h+s; s>=8 -> 0.
            union { v16h v; uint32_t u[8]; } bf;
            #pragma unroll
            for (int j = 0; j < 8; ++j)
                bf.u[j] = (j < 4) ? *(const uint32_t*)&wnT[p][lm][hi * 8 + 2 * j] : 0u;

            #pragma unroll
            for (int ct = 0; ct < 5; ++ct) {
                const int c = ct * 16 + lm;        // A M-row = c channel
                union { v16h v; uint32_t u[8]; } af;
                #pragma unroll
                for (int q = 0; q < 4; ++q) {
                    const int k0 = (hi ? 8 : 0) + 2 * q;   // A 16x32 layout, K<16
                    af.u[q]     = *(const uint32_t*)&featT[p][c][k0];
                    af.u[q + 4] = 0u;                       // K = 16..31 zero pad
                }
                v8f acc = {};
                acc = __builtin_amdgcn_wmma_f32_16x16x32_f16(
                        false, af.v, false, bf.v, (short)0, acc, false, false);
                #pragma unroll
                for (int r = 0; r < 8; ++r) {
                    const int cc = ct * 16 + r + (hi ? 8 : 0);
                    const int f  = cc * C_MID + lm;
                    if (f < F_PAD) pconv[p][f] = (_Float16)acc[r];
                }
            }
        }
    }
    __syncthreads();

    // ------- Phase 3: out[p,o] = sum_f pconv[p,f]*W[o,f] + b[o]  (WMMA GEMM) -------
    {
        const int hi = lane >> 4;
        const int lm = lane & 15;
        const int o0 = wv * 16;                    // this wave's output tile
        const v16h* wptr = (const v16h*)Wpk;
        v8f acc = {};
        for (int kt = 0; kt < NKT; ++kt) {
            union { v16h v; uint32_t u[8]; } af;   // A: M=point, K=f
            #pragma unroll
            for (int q = 0; q < 8; ++q) {
                const int f = kt * 32 + ((q < 4) ? 0 : 16) + hi * 8 + (q & 3) * 2;
                af.u[q] = *(const uint32_t*)&pconv[lm][f];
            }
            const v16h bfv = wptr[(kt * NWAVE + wv) * 32 + lane];   // 32B per lane
            if (kt + 1 < NKT)                       // prefetch next B tile chunk
                __builtin_prefetch(wptr + ((kt + 1) * NWAVE + wv) * 32 + lane, 0, 3);
            acc = __builtin_amdgcn_wmma_f32_16x16x32_f16(
                    false, af.v, false, bfv, (short)0, acc, false, false);
        }
        const float bsv = bias[o0 + lm];
        #pragma unroll
        for (int r = 0; r < 8; ++r) {
            const int pl = r + hi * 8;             // D M-row = point in tile
            const size_t g = (size_t)(g0 + pl);
            out[g * C_OUT + o0 + lm] = acc[r] + bsv;
        }
    }
}

// ---------------------------------------------------------------------------
extern "C" void kernel_launch(void* const* d_in, const int* in_sizes, int n_in,
                              void* d_out, int out_size, void* d_ws, size_t ws_size,
                              hipStream_t stream) {
    const float*     input_feat = (const float*)d_in[0];
    const long long* ninds      = (const long long*)d_in[1];
    const float*     weightnet  = (const float*)d_in[2];
    const float*     addfeat    = (const float*)d_in[3];
    const float*     lw         = (const float*)d_in[4];
    const float*     lb         = (const float*)d_in[5];
    float*           out        = (float*)d_out;
    uint32_t*        wpk        = (uint32_t*)d_ws;   // needs NKT*8*32*8*4 = 278528 B

    // Repack W into WMMA B-fragment order (f16), once per call.
    const int total_uints = NKT * 8 * 32 * 8;        // 69632
    prep_weights<<<(total_uints + 255) / 256, 256, 0, stream>>>(lw, wpk);

    // Fused gather + double-WMMA kernel: 80000 points / 16 per WG = 5000 WGs.
    pconv_fused<<<(BB * NN) / PTS, 256, 0, stream>>>(
        input_feat, ninds, weightnet, addfeat, lb, wpk, out);
}
